// FlowGuidedBlock_39204461478389
// MI455X (gfx1250) — compile-verified, matrix operand order checked by reading, not run on hardware
//
#include <hip/hip_runtime.h>
#include <hip/hip_bf16.h>

// ---------------------------------------------------------------------------
// Problem constants: B=4, C=64, H=W=160, OUT=64
// ---------------------------------------------------------------------------
#define Bn   4
#define Cn   64
#define Hn   160
#define Wn   160
#define HWn  (Hn * Wn)          // 25600
#define NPIX (Bn * HWn)         // 102400

typedef __attribute__((ext_vector_type(16))) __bf16 v16bf;
typedef __attribute__((ext_vector_type(8)))  __bf16 v8bf;
typedef __attribute__((ext_vector_type(8)))  float  v8f;

// 32-bit LDS byte offset of a shared-memory pointer (flat LDS aperture keeps
// the workgroup-relative offset in addr[31:0]).
__device__ __forceinline__ unsigned lds_off(const void* p) {
    return (unsigned)(unsigned long long)p;
}

// CDNA5 async global->LDS copy (ASYNCcnt). vdst = LDS byte address VGPR,
// vaddr = 64-bit global address VGPR pair, GV mode.
__device__ __forceinline__ void async_b128(unsigned lds, const void* g) {
    asm volatile("global_load_async_to_lds_b128 %0, %1, off"
                 :: "v"(lds), "v"(g) : "memory");
}
__device__ __forceinline__ void wait_async0() {
    asm volatile("s_wait_asynccnt 0" ::: "memory");
}

// Build a 16-element bf16 fragment from two contiguous 8-element LDS runs
// (each lowers to one ds_load_b128).
__device__ __forceinline__ v16bf frag16(const __bf16* lo, const __bf16* hi) {
    v8bf a = *(const v8bf*)lo;
    v8bf b = *(const v8bf*)hi;
    return __builtin_shufflevector(a, b, 0, 1, 2, 3, 4, 5, 6, 7,
                                   8, 9, 10, 11, 12, 13, 14, 15);
}

// ---------------------------------------------------------------------------
// Sobel edge map, averaged over channels. grid=(Hn,Bn), block=Wn (5 waves).
// ---------------------------------------------------------------------------
__global__ void edge_kernel(const float* __restrict__ x, float* __restrict__ edge) {
    const int w = threadIdx.x;
    const int h = blockIdx.x;
    const int b = blockIdx.y;
    float acc = 0.f;
    for (int c = 0; c < Cn; ++c) {
        const float* base = x + ((size_t)(b * Cn + c)) * HWn;
        float v[3][3];
#pragma unroll
        for (int dh = 0; dh < 3; ++dh)
#pragma unroll
            for (int dw = 0; dw < 3; ++dw) {
                int hh = h + dh - 1, ww = w + dw - 1;
                v[dh][dw] = ((unsigned)hh < (unsigned)Hn && (unsigned)ww < (unsigned)Wn)
                                ? base[hh * Wn + ww] : 0.f;
            }
        float gx = -v[0][0] + v[0][2] - 2.f * v[1][0] + 2.f * v[1][2] - v[2][0] + v[2][2];
        float gy = -v[0][0] - 2.f * v[0][1] - v[0][2] + v[2][0] + 2.f * v[2][1] + v[2][2];
        acc += fabsf(gx) + fabsf(gy);
    }
    edge[(size_t)(b * Hn + h) * Wn + w] = acc * (1.f / (float)Cn);
}

// ---------------------------------------------------------------------------
// x (fp32 NCHW) -> xt (bf16 pixel-major [NPIX][64]).
// grid=(Wn*64/256, Hn, Bn), block=256. i = w*64+c within the row.
// ---------------------------------------------------------------------------
__global__ void xprep_kernel(const float* __restrict__ x, __bf16* __restrict__ xt) {
    const int i = blockIdx.x * 256 + threadIdx.x;   // [0, Wn*64)
    const int w = i >> 6, c = i & 63;
    const int h = blockIdx.y;
    const int b = blockIdx.z;
    xt[((size_t)((b * Hn + h) * Wn) + w) * 64 + c] =
        (__bf16)x[((size_t)(b * Cn + c) * Hn + h) * Wn + w];
}

// ---------------------------------------------------------------------------
// OIHW fp32 conv weights -> [tap][O][C] bf16 (contiguous per tap for DMA).
// ---------------------------------------------------------------------------
__global__ void wtap_prep_kernel(const float* __restrict__ src, __bf16* __restrict__ dst,
                                 int cout) {
    int i = blockIdx.x * blockDim.x + threadIdx.x;
    int total = 9 * cout * 64;
    if (i >= total) return;
    int tap = i / (cout * 64);
    int rem = i - tap * cout * 64;
    int o = rem >> 6, c = rem & 63;
    dst[i] = (__bf16)src[(o * 64 + c) * 9 + tap];
}

// ---------------------------------------------------------------------------
// Reduce dyn1d kernels over K -> w_eff (64x64 bf16) + b_eff; convert fc_w.
// concat order: [rk5(16), rk7(16), lk5(16), lk7(16)]
// ---------------------------------------------------------------------------
__global__ void weff_kernel(const float* __restrict__ rk5w, const float* __restrict__ rk5b,
                            const float* __restrict__ rk7w, const float* __restrict__ rk7b,
                            const float* __restrict__ lk5w, const float* __restrict__ lk5b,
                            const float* __restrict__ lk7w, const float* __restrict__ lk7b,
                            const float* __restrict__ fcw,
                            __bf16* __restrict__ w_eff, float* __restrict__ b_eff,
                            __bf16* __restrict__ w_fc) {
    int i = blockIdx.x * blockDim.x + threadIdx.x;
    if (i >= 64 * 64) return;
    int o = i >> 6, c = i & 63;
    int g = o >> 4, oo = o & 15;
    int K = (g & 1) ? 7 : 5;
    const float* wp = (g == 0) ? rk5w : (g == 1) ? rk7w : (g == 2) ? lk5w : lk7w;
    const float* bp = (g == 0) ? rk5b : (g == 1) ? rk7b : (g == 2) ? lk5b : lk7b;
    float s = 0.f, sb = 0.f;
    for (int k = 0; k < K; ++k) {
        s  += wp[(oo * K + k) * 64 + c];
        sb += bp[oo * K + k];
    }
    w_eff[i] = (__bf16)s;
    if (c == 0) b_eff[o] = sb;
    w_fc[i] = (__bf16)fcw[i];
}

// ---------------------------------------------------------------------------
// WMMA 3x3 conv as 9 shifted 1x1 GEMMs, double-buffered async DMA pipeline.
// Row-aligned pixel tiles: grid=(Wn/MPIX, Hn, Bn) -> no integer division.
// Per tap: issue async DMA for tap+1 into buf^1, compute tap from buf
// (ds_load_b128 fragments + 2 WMMA), s_wait_asynccnt 0, one barrier.
// MODE 0: out = relu(v + bias)        MODE 1: out = relu(bn(v + bias))
// ---------------------------------------------------------------------------
template <int COUT, int MODE>
__global__ __launch_bounds__(128) void conv3x3_wmma_kernel(
    const __bf16* __restrict__ in_t,   // [NPIX][64] pixel-major
    const __bf16* __restrict__ wtap,   // [9][COUT][64]
    const float* __restrict__ bias,
    const float* __restrict__ bng, const float* __restrict__ bnb,
    const float* __restrict__ bnm, const float* __restrict__ bnv,
    __bf16* __restrict__ out_t)        // [NPIX][COUT]
{
    constexpr int NT   = COUT / 16;
    constexpr int MT   = 4 / NT;
    constexpr int MPIX = 16 * MT;

    __shared__ __attribute__((aligned(16))) __bf16 As[2][MPIX][64];
    __shared__ __attribute__((aligned(16))) __bf16 Bs[2][COUT][64];

    const int tid  = threadIdx.x;
    const int lane = tid & 31;
    const int wave = tid >> 5;
    const int mt   = wave / NT;
    const int nt   = wave - mt * NT;

    const int wBase = blockIdx.x * MPIX;       // tile start within row
    const int h     = blockIdx.y;
    const int b     = blockIdx.z;
    const int rowBase = (b * Hn + h) * Wn;     // flat pixel index of row start

    const int kb   = (lane >> 4) << 3;         // 0 or 8
    const int mrow = (lane & 15) + mt * 16;
    const int ncol = (lane & 15) + nt * 16;

    const unsigned asb = lds_off(&As[0][0][0]);
    const unsigned bsb = lds_off(&Bs[0][0][0]);

    auto stage = [&](int tap, int bufi) {
        const int dh = tap / 3 - 1;
        const int dw = tap - (tap / 3) * 3 - 1;
        const int hh = h + dh;
        const bool hok = (unsigned)hh < (unsigned)Hn;
        const size_t grow = (size_t)(b * Hn + (hok ? hh : 0)) * Wn;
        // A tile: MPIX rows x 128B from horizontally shifted pixels
#pragma unroll
        for (int i = tid; i < MPIX * 8; i += 128) {
            int m = i >> 3, part = i & 7;
            int ww = wBase + m + dw;
            unsigned la = asb + (unsigned)(bufi * MPIX * 128 + m * 128 + part * 16);
            if (hok && (unsigned)ww < (unsigned)Wn) {
                const char* g = (const char*)in_t + (grow + ww) * 128 + part * 16;
                async_b128(la, g);
            } else {
                *(uint4*)&As[bufi][m][part * 8] = make_uint4(0u, 0u, 0u, 0u);
            }
        }
        // B tile: COUT rows x 128B, contiguous per tap
        const char* g0 = (const char*)wtap + (size_t)tap * COUT * 128;
#pragma unroll
        for (int i = tid; i < COUT * 8; i += 128)
            async_b128(bsb + (unsigned)(bufi * COUT * 128 + i * 16), g0 + i * 16);
    };

    v8f acc = {0.f, 0.f, 0.f, 0.f, 0.f, 0.f, 0.f, 0.f};

    stage(0, 0);
    wait_async0();
    __syncthreads();

    int buf = 0;
    for (int tap = 0; tap < 9; ++tap) {
        if (tap < 8) stage(tap + 1, buf ^ 1);   // DMA next tap while computing
#pragma unroll
        for (int k0 = 0; k0 < 64; k0 += 32) {
            v16bf af  = frag16(&As[buf][mrow][k0 + kb], &As[buf][mrow][k0 + 16 + kb]);
            v16bf bfr = frag16(&Bs[buf][ncol][k0 + kb], &Bs[buf][ncol][k0 + 16 + kb]);
            acc = __builtin_amdgcn_wmma_f32_16x16x32_bf16(
                false, af, false, bfr, (short)0, acc, false, false);
        }
        wait_async0();
        __syncthreads();
        buf ^= 1;
    }

    // epilogue: pixel-major coalesced stores, division-free indexing
    float bval = bias[ncol];
    float sc = 1.f, sh = 0.f;
    if constexpr (MODE == 1) {
        sc = bng[ncol] * rsqrtf(bnv[ncol] + 1e-5f);
        sh = bnb[ncol] - bnm[ncol] * sc;
    }
#pragma unroll
    for (int r = 0; r < 8; ++r) {
        int m = r + ((lane >> 4) << 3) + mt * 16;
        float v = acc[r] + bval;
        if constexpr (MODE == 1) v = v * sc + sh;
        v = fmaxf(v, 0.f);
        out_t[(size_t)(rowBase + wBase + m) * COUT + ncol] = (__bf16)v;
    }
}

// ---------------------------------------------------------------------------
// Attention combine: a2 = <a1,ba_w2>+ba_b2; a = sigmoid(a2+edge);
// xa = x*(1+a) -> pixel-major bf16 + per-pixel channel sum.
// grid=(Hn,Bn), block=Wn.
// ---------------------------------------------------------------------------
__global__ void attn_kernel(const float* __restrict__ x, const __bf16* __restrict__ a1,
                            const float* __restrict__ ba_w2, const float* __restrict__ ba_b2,
                            const float* __restrict__ edge,
                            __bf16* __restrict__ xa, float* __restrict__ xsum)
{
    const int w = threadIdx.x;
    const int h = blockIdx.x;
    const int b = blockIdx.y;
    const int hw = h * Wn + w;
    const size_t p = (size_t)b * HWn + hw;
    float s = 0.f;
    for (int c = 0; c < 32; ++c)
        s += (float)a1[p * 32 + c] * ba_w2[c];
    float a = 1.f / (1.f + expf(-(s + ba_b2[0] + edge[p])));
    float f = 1.f + a;
    float tot = 0.f;
    for (int c = 0; c < Cn; ++c) {
        float v = x[((size_t)(b * Cn + c)) * HWn + hw] * f;
        tot += v;
        xa[p * Cn + c] = (__bf16)v;
    }
    xsum[p] = tot;
}

// ---------------------------------------------------------------------------
// WMMA 1x1 conv: M = 16 pixels/block, N = 64, K = 64. Whole K staged once via
// async DMA (A: 2KB, B: 8KB), one wait + one barrier, two WMMA steps.
// Row-aligned tiles: grid=(Wn/16, Hn, Bn).
// MODE 0 (dyn1d): out_t = (v + bias) * xsum[p]   (pixel-major bf16)
// MODE 1 (fc):    outf  = v + bias + resid       (NCHW fp32, final output)
// ---------------------------------------------------------------------------
template <int MODE>
__global__ __launch_bounds__(128) void conv1x1_wmma_kernel(
    const __bf16* __restrict__ in_t,   // [NPIX][64]
    const __bf16* __restrict__ wbf,    // [64][64]
    const float*  __restrict__ bias,   // (64)
    const float*  __restrict__ xsum,   // MODE 0
    const float*  __restrict__ resid,  // MODE 1: x (fp32 NCHW)
    __bf16* __restrict__ out_t,
    float*  __restrict__ outf)
{
    __shared__ __attribute__((aligned(16))) __bf16 As[16][64];
    __shared__ __attribute__((aligned(16))) __bf16 Bs[64][64];

    const int tid  = threadIdx.x;
    const int lane = tid & 31;
    const int wave = tid >> 5;                 // NT = 4
    const int wBase = blockIdx.x * 16;
    const int h     = blockIdx.y;
    const int b     = blockIdx.z;
    const int pixBase = (b * Hn + h) * Wn + wBase;

    const int kb   = (lane >> 4) << 3;
    const int mrow = lane & 15;
    const int ncol = (lane & 15) + (wave << 4);

    const unsigned asb = lds_off(&As[0][0]);
    const unsigned bsb = lds_off(&Bs[0][0]);

    // stage A (16x128B contiguous) and B (64x128B contiguous)
    {
        const char* ga = (const char*)in_t + (size_t)pixBase * 128;
#pragma unroll
        for (int i = tid; i < 16 * 8; i += 128)        // 1 per thread
            async_b128(asb + (unsigned)(i * 16), ga + i * 16);
#pragma unroll
        for (int i = tid; i < 64 * 8; i += 128)        // 4 per thread
            async_b128(bsb + (unsigned)(i * 16), (const char*)wbf + i * 16);
    }
    wait_async0();
    __syncthreads();

    v8f acc = {0.f, 0.f, 0.f, 0.f, 0.f, 0.f, 0.f, 0.f};
#pragma unroll
    for (int k0 = 0; k0 < 64; k0 += 32) {
        v16bf af  = frag16(&As[mrow][k0 + kb], &As[mrow][k0 + 16 + kb]);
        v16bf bfr = frag16(&Bs[ncol][k0 + kb], &Bs[ncol][k0 + 16 + kb]);
        acc = __builtin_amdgcn_wmma_f32_16x16x32_bf16(
            false, af, false, bfr, (short)0, acc, false, false);
    }

    float bval = bias[ncol];
#pragma unroll
    for (int r = 0; r < 8; ++r) {
        int m = r + ((lane >> 4) << 3);
        float v = acc[r] + bval;
        if constexpr (MODE == 0) {
            v *= xsum[pixBase + m];
            out_t[(size_t)(pixBase + m) * 64 + ncol] = (__bf16)v;
        } else {
            size_t oidx = ((size_t)(b * 64 + ncol) * Hn + h) * Wn + (wBase + m);
            v += resid[oidx];
            outf[oidx] = v;
        }
    }
}

// ---------------------------------------------------------------------------
// Host-side launch. Workspace layout (offsets in bytes):
//   edge f32 @0MB, xsum f32 @1MB,
//   w_eff bf16 @2MB, w_fc bf16 @2MB+16KB, b_eff f32 @2MB+32KB,
//   wtap_ba1 @2MB+64KB (36KB), wtap_sf @2MB+128KB (72KB),
//   xt @4MB (12.5MB), a1 @17MB (6.25MB), xa @24MB (12.5MB),
//   fusion @37MB (12.5MB), tbuf @50MB (12.5MB).  Total < 63MB.
// ---------------------------------------------------------------------------
#define WS_MB (1024ull * 1024ull)
#define WS_KB (1024ull)

extern "C" void kernel_launch(void* const* d_in, const int* in_sizes, int n_in,
                              void* d_out, int out_size, void* d_ws, size_t ws_size,
                              hipStream_t stream)
{
    const float* x       = (const float*)d_in[0];
    const float* ba_w1   = (const float*)d_in[1];
    const float* ba_b1   = (const float*)d_in[2];
    const float* ba_w2   = (const float*)d_in[3];
    const float* ba_b2   = (const float*)d_in[4];
    // d_in[5..12] = off_* path: dead in the reference graph (DCE'd).
    const float* rk5_w   = (const float*)d_in[13];
    const float* rk5_b   = (const float*)d_in[14];
    const float* rk7_w   = (const float*)d_in[15];
    const float* rk7_b   = (const float*)d_in[16];
    const float* lk5_w   = (const float*)d_in[17];
    const float* lk5_b   = (const float*)d_in[18];
    const float* lk7_w   = (const float*)d_in[19];
    const float* lk7_b   = (const float*)d_in[20];
    const float* sf_w    = (const float*)d_in[21];
    const float* sf_b    = (const float*)d_in[22];
    const float* sf_bn_g = (const float*)d_in[23];
    const float* sf_bn_b = (const float*)d_in[24];
    const float* sf_bn_m = (const float*)d_in[25];
    const float* sf_bn_v = (const float*)d_in[26];
    const float* fc_w    = (const float*)d_in[27];
    const float* fc_b    = (const float*)d_in[28];

    char* ws = (char*)d_ws;
    float*  edge     = (float*) (ws + 0 * WS_MB);
    float*  xsum     = (float*) (ws + 1 * WS_MB);
    __bf16* w_eff    = (__bf16*)(ws + 2 * WS_MB);
    __bf16* w_fc     = (__bf16*)(ws + 2 * WS_MB + 16 * WS_KB);
    float*  b_eff    = (float*) (ws + 2 * WS_MB + 32 * WS_KB);
    __bf16* wtap_ba1 = (__bf16*)(ws + 2 * WS_MB + 64 * WS_KB);
    __bf16* wtap_sf  = (__bf16*)(ws + 2 * WS_MB + 128 * WS_KB);
    __bf16* xt       = (__bf16*)(ws + 4 * WS_MB);
    __bf16* a1       = (__bf16*)(ws + 17 * WS_MB);
    __bf16* xa       = (__bf16*)(ws + 24 * WS_MB);
    __bf16* fusion   = (__bf16*)(ws + 37 * WS_MB);
    __bf16* tbuf     = (__bf16*)(ws + 50 * WS_MB);

    // 1) Sobel edge map
    edge_kernel<<<dim3(Hn, Bn), Wn, 0, stream>>>(x, edge);

    // 2) prep: x -> pixel-major bf16; weights -> tap-major bf16; w_eff/w_fc
    xprep_kernel<<<dim3((Wn * 64) / 256, Hn, Bn), 256, 0, stream>>>(x, xt);
    wtap_prep_kernel<<<(9 * 32 * 64 + 255) / 256, 256, 0, stream>>>(ba_w1, wtap_ba1, 32);
    wtap_prep_kernel<<<(9 * 64 * 64 + 255) / 256, 256, 0, stream>>>(sf_w, wtap_sf, 64);
    weff_kernel<<<(64 * 64) / 256, 256, 0, stream>>>(
        rk5_w, rk5_b, rk7_w, rk7_b, lk5_w, lk5_b, lk7_w, lk7_b, fc_w,
        w_eff, b_eff, w_fc);

    // 3) boundary_attn conv1: 3x3, 64->32, relu (WMMA, MPIX=32 row tiles)
    conv3x3_wmma_kernel<32, 0><<<dim3(Wn / 32, Hn, Bn), 128, 0, stream>>>(
        xt, wtap_ba1, ba_b1, nullptr, nullptr, nullptr, nullptr, a1);

    // 4) 1x1 32->1 + sigmoid gate -> xa (pixel-major bf16) + channel sum
    attn_kernel<<<dim3(Hn, Bn), Wn, 0, stream>>>(x, a1, ba_w2, ba_b2, edge, xa, xsum);

    // 5) fusion = xsum * (1x1(xa, w_eff) + b_eff)   (WMMA, async staged)
    conv1x1_wmma_kernel<0><<<dim3(Wn / 16, Hn, Bn), 128, 0, stream>>>(
        xa, w_eff, b_eff, xsum, nullptr, fusion, nullptr);

    // 6) sf: 3x3 64->64 + BN + ReLU   (WMMA, double-buffered async pipeline)
    conv3x3_wmma_kernel<64, 1><<<dim3(Wn / 16, Hn, Bn), 128, 0, stream>>>(
        fusion, wtap_sf, sf_b, sf_bn_g, sf_bn_b, sf_bn_m, sf_bn_v, tbuf);

    // 7) fc: 1x1 64->64 + bias + residual -> fp32 NCHW output
    conv1x1_wmma_kernel<1><<<dim3(Wn / 16, Hn, Bn), 128, 0, stream>>>(
        tbuf, w_fc, fc_b, nullptr, x, nullptr, (float*)d_out);
}
